// GAT_88201448391435
// MI455X (gfx1250) — matrix operand order repaired and checked by reading, not compiled, not run on hardware
//
#include <hip/hip_runtime.h>
#include <hip/hip_bf16.h>
#include <math.h>

typedef __attribute__((ext_vector_type(16))) _Float16 v16h;
typedef __attribute__((ext_vector_type(8)))  float    v8f;

#define NODES   50000
#define E0      800000
#define ET      850000   /* E0 + self loops */
#define HID     128
#define HEADS   4
#define HC1     512      /* HEADS*HID */
#define NEG_SLOPE 0.2f

// ---------------------------------------------------------------------------
// Repack weight matrix W[K,N] (f32, row major) into fragment-major f16:
// P[((ntile*(K/32) + kblk)*32 + lane)*16 + e]  ==  B-fragment element e of
// lane `lane` for output tile column `ntile`, K-block `kblk`, following the
// contiguous k order of the A side (lanes0-15: K=2i+p, lanes16-31: +16).
// Each lane's 16 halfs are then one contiguous 32B load in the GEMM.
// ---------------------------------------------------------------------------
__global__ void pack_weights(const float* __restrict__ W, _Float16* __restrict__ P,
                             int N, int K)
{
    const int t = (int)(blockIdx.x * (size_t)blockDim.x + threadIdx.x);
    if (t >= N * K) return;
    const int kblocks = K >> 5;
    const int e     = t & 15;
    const int lane  = (t >> 4) & 31;
    const int kblk  = (t >> 9) % kblocks;
    const int ntile = (t >> 9) / kblocks;
    const int h  = lane >> 4;
    const int n  = ntile * 16 + (lane & 15);
    const int kb = 2 * (e >> 1) + (e & 1) + 16 * h;
    P[t] = (_Float16)W[(size_t)(kblk * 32 + kb) * N + n];
}

// ---------------------------------------------------------------------------
// WMMA GEMM: C[M,N] = A[M,K] * Bpacked, f32 A/C, f16 multiply, f32 accumulate
// via v_wmma_f32_16x16x32_f16. One wave per 16x16 C tile. A is converted to
// f16 in registers (one contiguous 16-float run per lane -> b128 loads);
// B fragments come from the packed buffer as one 32B contiguous load/lane.
// ---------------------------------------------------------------------------
template <int N, int K>
__global__ __launch_bounds__(256) void gemm_wmma_f16(
    const float* __restrict__ A, const _Float16* __restrict__ pB,
    float* __restrict__ C, int M)
{
    const int wave  = threadIdx.x >> 5;
    const int lane  = threadIdx.x & 31;
    const int ntile = blockIdx.x;                   // N/16 tiles
    const int mtile = blockIdx.y * 8 + wave;        // 8 waves per block
    if (mtile * 16 >= M) return;                    // wave-uniform exit

    const int h   = lane >> 4;                      // half-wave select
    const int mn  = lane & 15;
    const int row = mtile * 16 + mn;                // A row for this lane
    const int col = ntile * 16 + mn;                // C col for this lane

    const v16h* __restrict__ bfrag =
        (const v16h*)pB + (size_t)(ntile * (K >> 5)) * 32 + lane;

    v8f acc = {};
#pragma unroll 4
    for (int kblk = 0; kblk < (K >> 5); ++kblk) {
        const int k0 = kblk * 32;
        v16h a;
#pragma unroll
        for (int e = 0; e < 16; ++e) {
            const int i = e >> 1, p = e & 1;
            // A 16x32 f16 layout (matches packed B): lanes0-15: K=0..15;
            //                                        lanes16-31: K=16..31
            const int ka = 2 * i + p + 16 * h;
            a[e] = (_Float16)A[(size_t)row * K + k0 + ka];
        }
        const v16h b = bfrag[(size_t)kblk * 32];
        acc = __builtin_amdgcn_wmma_f32_16x16x32_f16(
            false, a, false, b, (short)0, acc, false, false);
    }
#pragma unroll
    for (int r = 0; r < 8; ++r) {
        const int m = mtile * 16 + r + 8 * h;       // C/D: VGPR r -> M=r+8h
        C[(size_t)m * N + col] = acc[r];
    }
}

// ---------------------------------------------------------------------------
// alpha_s / alpha_d: per (node, head) dot products, one wave each (wave32)
// ---------------------------------------------------------------------------
__global__ void alpha_kernel(const float* __restrict__ xh,
                             const float* __restrict__ a_src,
                             const float* __restrict__ a_dst,
                             float* __restrict__ as, float* __restrict__ ad,
                             int n_nodes, int H)
{
    const int wid  = (int)((blockIdx.x * (size_t)blockDim.x + threadIdx.x) >> 5);
    const int lane = threadIdx.x & 31;
    if (wid >= n_nodes * H) return;
    const int n = wid / H, h = wid % H;
    const float* xr  = xh + (size_t)n * H * HID + (size_t)h * HID;
    const float* asr = a_src + h * HID;
    const float* adr = a_dst + h * HID;
    float s = 0.f, d = 0.f;
    for (int c = lane; c < HID; c += 32) {
        const float xv = xr[c];
        s += xv * asr[c];
        d += xv * adr[c];
    }
    for (int off = 16; off; off >>= 1) {
        s += __shfl_xor(s, off, 32);
        d += __shfl_xor(d, off, 32);
    }
    if (lane == 0) { as[wid] = s; ad[wid] = d; }
}

// ---------------------------------------------------------------------------
// helpers
// ---------------------------------------------------------------------------
__device__ __forceinline__ void atomicMaxF(float* addr, float v) {
    if (v >= 0.f) atomicMax((int*)addr, __float_as_int(v));
    else          atomicMin((unsigned int*)addr, __float_as_uint(v));
}

__device__ __forceinline__ void edge_ends(const long long* __restrict__ ei,
                                          int e, int& src, int& dst) {
    if (e < E0) { src = (int)ei[e]; dst = (int)ei[E0 + e]; }
    else        { src = dst = e - E0; }               // self loop
}

__global__ void fill_kernel(float* __restrict__ p, float v, size_t n) {
    size_t i = blockIdx.x * (size_t)blockDim.x + threadIdx.x;
    if (i < n) p[i] = v;
}

// ---------------------------------------------------------------------------
// Pass A: e = leakyrelu(as[src]+ad[dst]); stash raw logit; segment max (atomic)
// ---------------------------------------------------------------------------
__global__ void edge_logits(const float* __restrict__ as,
                            const float* __restrict__ ad,
                            const long long* __restrict__ ei,
                            float* __restrict__ ebuf, float* __restrict__ mbuf,
                            int H)
{
    const int e = (int)(blockIdx.x * (size_t)blockDim.x + threadIdx.x);
    if (e >= ET) return;
    int src, dst; edge_ends(ei, e, src, dst);
    for (int h = 0; h < H; ++h) {
        float v = as[src * H + h] + ad[dst * H + h];
        v = v > 0.f ? v : NEG_SLOPE * v;
        ebuf[(size_t)e * H + h] = v;
        atomicMaxF(&mbuf[dst * H + h], v);
    }
}

// ---------------------------------------------------------------------------
// Pass B: ex = exp(e - m[dst]); denom[dst] += ex (atomic); stash ex
// ---------------------------------------------------------------------------
__global__ void edge_exp(const long long* __restrict__ ei,
                         float* __restrict__ ebuf,
                         const float* __restrict__ mbuf,
                         float* __restrict__ denom, int H)
{
    const int e = (int)(blockIdx.x * (size_t)blockDim.x + threadIdx.x);
    if (e >= ET) return;
    int src, dst; edge_ends(ei, e, src, dst);
    (void)src;
    for (int h = 0; h < H; ++h) {
        const float v = expf(ebuf[(size_t)e * H + h] - mbuf[dst * H + h]);
        ebuf[(size_t)e * H + h] = v;
        atomicAdd(&denom[dst * H + h], v);
    }
}

// ---------------------------------------------------------------------------
// Pass C: out[dst] += xh[src] * (ex/denom[dst]); one 128-thread block per edge,
// fully coalesced across channels; f32 global atomics (L2-resident targets).
// ---------------------------------------------------------------------------
__global__ __launch_bounds__(128) void edge_aggregate(
    const long long* __restrict__ ei, const float* __restrict__ ebuf,
    const float* __restrict__ denom, const float* __restrict__ xh,
    float* __restrict__ out, int HC, int H)
{
    const int e = blockIdx.x;
    int src, dst; edge_ends(ei, e, src, dst);
    for (int jj = threadIdx.x; jj < HC; jj += 128) {
        const int h = jj >> 7;                        // C == 128 per head
        const float w = ebuf[(size_t)e * H + h] / denom[dst * H + h];
        atomicAdd(&out[(size_t)dst * HC + jj],
                  xh[(size_t)src * HC + jj] * w);
    }
}

// ---------------------------------------------------------------------------
// out = elu(out + bias) elementwise
// ---------------------------------------------------------------------------
__global__ void bias_elu(float* __restrict__ buf, const float* __restrict__ bias,
                         size_t total, int HC)
{
    size_t i = blockIdx.x * (size_t)blockDim.x + threadIdx.x;
    if (i >= total) return;
    const float v = buf[i] + bias[i % HC];
    buf[i] = v > 0.f ? v : expm1f(v);
}

// ---------------------------------------------------------------------------
// final head: out[n] = dot(out2[n] + b2, fc_w) + fc_b  (one wave per node)
// ---------------------------------------------------------------------------
__global__ void final_fc(const float* __restrict__ out2,
                         const float* __restrict__ b2,
                         const float* __restrict__ fc_w,
                         const float* __restrict__ fc_b,
                         float* __restrict__ out, int n_nodes)
{
    const int wid  = (int)((blockIdx.x * (size_t)blockDim.x + threadIdx.x) >> 5);
    const int lane = threadIdx.x & 31;
    if (wid >= n_nodes) return;
    const float* r = out2 + (size_t)wid * HID;
    float s = 0.f;
    for (int c = lane; c < HID; c += 32) s += (r[c] + b2[c]) * fc_w[c];
    for (int off = 16; off; off >>= 1) s += __shfl_xor(s, off, 32);
    if (lane == 0) out[wid] = s + fc_b[0];
}

// ---------------------------------------------------------------------------
extern "C" void kernel_launch(void* const* d_in, const int* in_sizes, int n_in,
                              void* d_out, int out_size, void* d_ws, size_t ws_size,
                              hipStream_t stream)
{
    (void)in_sizes; (void)n_in; (void)out_size; (void)ws_size;
    const float*      x      = (const float*)d_in[0];
    const long long*  ei     = (const long long*)d_in[1];   // int64 [2, E0]
    const float*      W1     = (const float*)d_in[2];
    const float*      a_src1 = (const float*)d_in[3];
    const float*      a_dst1 = (const float*)d_in[4];
    const float*      b1     = (const float*)d_in[5];
    const float*      W2     = (const float*)d_in[6];
    const float*      a_src2 = (const float*)d_in[7];
    const float*      a_dst2 = (const float*)d_in[8];
    const float*      b2     = (const float*)d_in[9];
    const float*      fc_w   = (const float*)d_in[10];
    const float*      fc_b   = (const float*)d_in[11];
    float*            out    = (float*)d_out;
    float*            ws     = (float*)d_ws;

    // ---- workspace layout (floats) -------------------------------------
    const size_t NHC1 = (size_t)NODES * HC1;           // 25.6M
    float* xh1  = ws;                                  // [N, 512]
    float* h1   = ws + NHC1;                           // [N, 512] out1 -> elu
    float* as1  = ws + 2 * NHC1;                       // [N, 4]
    float* ad1  = as1 + (size_t)NODES * HEADS;
    float* m1   = ad1 + (size_t)NODES * HEADS;
    float* den1 = m1  + (size_t)NODES * HEADS;
    float* ex1  = den1 + (size_t)NODES * HEADS;        // [ET, 4]
    _Float16* pW1 = (_Float16*)(ex1 + (size_t)ET * HEADS);   // 65536 halfs
    _Float16* pW2 = pW1 + (size_t)HC1 * HID;                 // 65536 halfs
    // layer-2 buffers reuse the (dead-after-layer-1) xh1 region
    float* xh2  = xh1;                                 // [N, 128]
    float* out2 = xh1 + (size_t)NODES * HID;           // [N, 128]
    float* as2  = xh1 + 2 * (size_t)NODES * HID;
    float* ad2  = as2 + NODES;
    float* m2   = ad2 + NODES;
    float* den2 = m2  + NODES;
    float* ex2  = den2 + NODES;                        // [ET]

    const dim3 blk256(256), blk128(128);
    const int egrid = (ET + 255) / 256;

    // ---- pack both weight matrices into WMMA fragment order (f16) ------
    pack_weights<<<(HC1 * 128 + 255) / 256, blk256, 0, stream>>>(W1, pW1, HC1, 128);
    pack_weights<<<(HID * HC1 + 255) / 256, blk256, 0, stream>>>(W2, pW2, HID, HC1);

    // ================= Layer 1: GAT(128 -> 4 x 128, concat) =============
    // xh1 = x @ W1  (M=50000, N=512, K=128)
    {
        dim3 grid(HC1 / 16, (NODES / 16 + 7) / 8);
        gemm_wmma_f16<HC1, 128><<<grid, blk256, 0, stream>>>(x, pW1, xh1, NODES);
    }
    // alpha dots
    alpha_kernel<<<(NODES * HEADS * 32 + 255) / 256, blk256, 0, stream>>>(
        xh1, a_src1, a_dst1, as1, ad1, NODES, HEADS);
    // segment softmax state
    fill_kernel<<<(NODES * HEADS + 255) / 256, blk256, 0, stream>>>(m1, -1e30f, (size_t)NODES * HEADS);
    fill_kernel<<<(NODES * HEADS + 255) / 256, blk256, 0, stream>>>(den1, 0.f, (size_t)NODES * HEADS);
    edge_logits<<<egrid, blk256, 0, stream>>>(as1, ad1, ei, ex1, m1, HEADS);
    edge_exp   <<<egrid, blk256, 0, stream>>>(ei, ex1, m1, den1, HEADS);
    // aggregate messages
    fill_kernel<<<(NHC1 + 255) / 256, blk256, 0, stream>>>(h1, 0.f, NHC1);
    edge_aggregate<<<ET, blk128, 0, stream>>>(ei, ex1, den1, xh1, h1, HC1, HEADS);
    // h1 = elu(h1 + b1)
    bias_elu<<<(NHC1 + 255) / 256, blk256, 0, stream>>>(h1, b1, NHC1, HC1);

    // ================= Layer 2: GAT(512 -> 128, 1 head) =================
    // xh2 = h1 @ W2  (M=50000, N=128, K=512)
    {
        dim3 grid(HID / 16, (NODES / 16 + 7) / 8);
        gemm_wmma_f16<HID, HC1><<<grid, blk256, 0, stream>>>(h1, pW2, xh2, NODES);
    }
    alpha_kernel<<<(NODES * 32 + 255) / 256, blk256, 0, stream>>>(
        xh2, a_src2, a_dst2, as2, ad2, NODES, 1);
    fill_kernel<<<(NODES + 255) / 256, blk256, 0, stream>>>(m2, -1e30f, (size_t)NODES);
    fill_kernel<<<(NODES + 255) / 256, blk256, 0, stream>>>(den2, 0.f, (size_t)NODES);
    edge_logits<<<egrid, blk256, 0, stream>>>(as2, ad2, ei, ex2, m2, 1);
    edge_exp   <<<egrid, blk256, 0, stream>>>(ei, ex2, m2, den2, 1);
    fill_kernel<<<((size_t)NODES * HID + 255) / 256, blk256, 0, stream>>>(out2, 0.f, (size_t)NODES * HID);
    edge_aggregate<<<ET, blk128, 0, stream>>>(ei, ex2, den2, xh2, out2, HID, 1);

    // ================= regression head ==================================
    final_fc<<<(NODES * 32 + 255) / 256, blk256, 0, stream>>>(out2, b2, fc_w, fc_b, out, NODES);
}
